// QuantLSTMLayer_10101763080415
// MI455X (gfx1250) — matrix operand order repaired and checked by reading, not compiled, not run on hardware
//
#include <hip/hip_runtime.h>

typedef __bf16 bf16;
typedef __attribute__((ext_vector_type(16))) __bf16 bf16x16;
typedef __attribute__((ext_vector_type(8)))  __bf16 bf16x8;
typedef __attribute__((ext_vector_type(4)))  __bf16 bf16x4;
typedef __attribute__((ext_vector_type(8)))  float  f32x8;
typedef __attribute__((ext_vector_type(4)))  float  f32x4;

// Problem sizes (fixed by the reference)
constexpr int   T_  = 512;
constexpr int   B_  = 64;
constexpr int   I_  = 1024;
constexpr int   H_  = 1024;
constexpr int   G4H = 4 * H_;           // 4096 gate columns

// ---------------- workspace layout (bytes) ----------------
constexpr size_t OFF_WI   = 0;                                   // bf16 [4H][I]    8 MB
constexpr size_t OFF_WH   = OFF_WI + (size_t)G4H * I_ * 2;       // bf16 [4H][H]    8 MB
constexpr size_t OFF_BIAS = OFF_WH + (size_t)G4H * H_ * 2;       // f32  [4H]       16 KB
constexpr size_t OFF_H0   = OFF_BIAS + (size_t)G4H * 4;          // bf16 [B][H]     128 KB
constexpr size_t OFF_H1   = OFF_H0 + (size_t)B_ * H_ * 2;        // bf16 [B][H]     128 KB
constexpr size_t OFF_C    = OFF_H1 + (size_t)B_ * H_ * 2;        // f32  [B][H]     256 KB
constexpr size_t OFF_XBF  = OFF_C  + (size_t)B_ * H_ * 4;        // bf16 [T*B][I]   64 MB
constexpr size_t OFF_XP   = OFF_XBF + (size_t)T_ * B_ * I_ * 2;  // f32  [T*B][4H]  512 MB

// ---------------- helpers ----------------
__device__ __forceinline__ float ht1(float x) {            // hardtanh [-1,1]
    return fminf(fmaxf(x, -1.0f), 1.0f);
}
__device__ __forceinline__ float sigm(float x) {
    return 1.0f / (1.0f + __expf(-x));
}
__device__ __forceinline__ float tanh_fast(float x) {
    float ax = fabsf(x);
    float e  = __expf(2.0f * ax);
    float t  = 1.0f - 2.0f / (e + 1.0f);
    return copysignf(t, x);
}

// A fragment: 16x32 bf16 tile, rows m0..m0+15, k-chunk k0 (contiguous k: lane half h holds k0+16h..+15)
__device__ __forceinline__ bf16x16 load_a_bf(const bf16* __restrict__ A, int ld,
                                             int m0, int k0, int lane) {
    int h = lane >> 4, m = lane & 15;
    const bf16* p = A + (size_t)(m0 + m) * ld + k0 + h * 16;
    bf16x8 c0 = *(const bf16x8*)p;          // K = k0 + h*16     .. +7
    bf16x8 c1 = *(const bf16x8*)(p + 8);    // K = k0 + h*16 + 8 .. +7
    bf16x16 r;
#pragma unroll
    for (int e = 0; e < 8; ++e) { r[e] = c0[e]; r[e + 8] = c1[e]; }
    return r;
}

// B fragment: B[k][n] = W[n][k]; cols n0..n0+15, k-chunk k0 (contiguous k: lane half h holds k0+16h..+15)
__device__ __forceinline__ bf16x16 load_b_frag(const bf16* __restrict__ W, int ld,
                                               int n0, int k0, int lane) {
    int h = lane >> 4, n = lane & 15;
    const bf16* p = W + (size_t)(n0 + n) * ld + k0 + h * 16;
    bf16x8 lo = *(const bf16x8*)p;
    bf16x8 hi = *(const bf16x8*)(p + 8);
    bf16x16 r;
#pragma unroll
    for (int e = 0; e < 8; ++e) { r[e] = lo[e]; r[e + 8] = hi[e]; }
    return r;
}

// ---------------- weight conversion: 4 fp32 gate blocks -> concat bf16 ----------------
__global__ void k_cvt4(const float* __restrict__ w0, const float* __restrict__ w1,
                       const float* __restrict__ w2, const float* __restrict__ w3,
                       bf16* __restrict__ dst, int rows_per_gate, int cols) {
    size_t total = (size_t)4 * rows_per_gate * cols;
    size_t block = (size_t)rows_per_gate * cols;
    for (size_t i = (size_t)blockIdx.x * blockDim.x + threadIdx.x; i < total;
         i += (size_t)gridDim.x * blockDim.x) {
        int g = (int)(i / block);
        size_t r = i - (size_t)g * block;
        const float* src = (g == 0) ? w0 : (g == 1) ? w1 : (g == 2) ? w2 : w3;
        dst[i] = (bf16)src[r];
    }
}

// ---------------- input quantization: Xbf = bf16(hardtanh(X)), vectorized x4 ----------------
__global__ void k_cvt_x(const float* __restrict__ X, bf16* __restrict__ Xbf) {
    size_t total4 = (size_t)T_ * B_ * I_ / 4;
    for (size_t i = (size_t)blockIdx.x * blockDim.x + threadIdx.x; i < total4;
         i += (size_t)gridDim.x * blockDim.x) {
        f32x4 v = *(const f32x4*)(X + i * 4);
        bf16x4 o;
#pragma unroll
        for (int e = 0; e < 4; ++e) o[e] = (bf16)ht1(v[e]);
        *(bf16x4*)(Xbf + i * 4) = o;
    }
}

// ---------------- state init: h0 = bf16(hardtanh(hx)), c = cx, bias concat ----------------
__global__ void k_init(const float* __restrict__ hx, const float* __restrict__ cx,
                       const float* __restrict__ b0, const float* __restrict__ b1,
                       const float* __restrict__ b2, const float* __restrict__ b3,
                       bf16* __restrict__ h_buf, float* __restrict__ c_ws,
                       float* __restrict__ bias) {
    for (size_t i = (size_t)blockIdx.x * blockDim.x + threadIdx.x; i < (size_t)B_ * H_;
         i += (size_t)gridDim.x * blockDim.x) {
        h_buf[i] = (bf16)ht1(hx[i]);
        c_ws[i]  = cx[i];
        if (i < (size_t)G4H) {
            int g = (int)(i / H_);
            size_t r = i - (size_t)g * H_;
            const float* src = (g == 0) ? b0 : (g == 1) ? b1 : (g == 2) ? b2 : b3;
            bias[i] = src[r];
        }
    }
}

// ---------------- big parallel input GEMM: Xp = Xbf @ Wi^T + bias ----------------
// Xbf: [T*B, I] bf16, Wi: [4H, I] bf16, Xp: [T*B, 4H] fp32
// grid = (4H/64, T*B/128), block = 256 (8 waves in 4(M) x 2(N)).
// Wave tile 32(M) x 32(N): 2 A frags + 2 B frags feed 4 WMMAs -> ~16 FLOP/B from cache.
__global__ void k_input_gemm(const bf16* __restrict__ Xbf, const bf16* __restrict__ Wi,
                             const float* __restrict__ bias, float* __restrict__ Xp) {
    int lane = threadIdx.x & 31;
    int wave = threadIdx.x >> 5;
    int wm = wave & 3;        // 0..3
    int wn = wave >> 2;       // 0..1
    int m0 = blockIdx.y * 128 + wm * 32;
    int n0 = blockIdx.x * 64 + wn * 32;

    f32x8 acc00 = {}, acc01 = {}, acc10 = {}, acc11 = {};
    for (int kc = 0; kc < I_; kc += 32) {
        bf16x16 a0 = load_a_bf(Xbf, I_, m0,      kc, lane);
        bf16x16 a1 = load_a_bf(Xbf, I_, m0 + 16, kc, lane);
        bf16x16 b0 = load_b_frag(Wi, I_, n0,      kc, lane);
        bf16x16 b1 = load_b_frag(Wi, I_, n0 + 16, kc, lane);
        acc00 = __builtin_amdgcn_wmma_f32_16x16x32_bf16(false, a0, false, b0, (short)0, acc00, false, false);
        acc01 = __builtin_amdgcn_wmma_f32_16x16x32_bf16(false, a0, false, b1, (short)0, acc01, false, false);
        acc10 = __builtin_amdgcn_wmma_f32_16x16x32_bf16(false, a1, false, b0, (short)0, acc10, false, false);
        acc11 = __builtin_amdgcn_wmma_f32_16x16x32_bf16(false, a1, false, b1, (short)0, acc11, false, false);
    }

    int h = lane >> 4, n = lane & 15;
    float bia0 = bias[n0 + n];
    float bia1 = bias[n0 + 16 + n];
#pragma unroll
    for (int r = 0; r < 8; ++r) {
        size_t row0 = (size_t)(m0 + h * 8 + r);
        size_t row1 = row0 + 16;
        Xp[row0 * G4H + n0 +  0 + n] = acc00[r] + bia0;
        Xp[row0 * G4H + n0 + 16 + n] = acc01[r] + bia1;
        Xp[row1 * G4H + n0 +  0 + n] = acc10[r] + bia0;
        Xp[row1 * G4H + n0 + 16 + n] = acc11[r] + bia1;
    }
}

// ---------------- one recurrent step ----------------
// grid = H/16 = 64 workgroups; block = 256 (8 waves).
// Workgroup owns hidden slice j0..j0+15 for ALL batch rows and ALL 4 gates:
//   wave w: batch tile bt = w&3 (rows 16*bt..), gate pair gp = w>>2 (gates 2gp, 2gp+1)
__global__ void k_lstm_step(const bf16* __restrict__ Wh, const float* __restrict__ Xp_t,
                            const bf16* __restrict__ h_in, bf16* __restrict__ h_out,
                            float* __restrict__ c_ws, float* __restrict__ out_t,
                            float* __restrict__ hn_out, float* __restrict__ cn_out,
                            int last) {
    __shared__ float lds_g[4][B_][16];   // 16 KB

    int lane = threadIdx.x & 31;
    int wave = threadIdx.x >> 5;
    int j0 = blockIdx.x * 16;

    int bt = wave & 3;
    int gp = wave >> 2;
    int m0 = bt * 16;
    int g0 = gp * 2, g1 = g0 + 1;
    int n0_0 = g0 * H_ + j0;
    int n0_1 = g1 * H_ + j0;

    // per-lane weight-row base pointers used for prefetch
    const bf16* pfr0 = Wh + (size_t)(n0_0 + (lane & 15)) * H_;
    const bf16* pfr1 = Wh + (size_t)(n0_1 + (lane & 15)) * H_;

    f32x8 acc0 = {}, acc1 = {};
    for (int kc = 0; kc < H_; kc += 32) {
        // pull W_h rows 8 chunks (256 bf16 = 512 B) ahead from L2 into near cache
        __builtin_prefetch(pfr0 + kc + 256, 0, 0);
        __builtin_prefetch(pfr1 + kc + 256, 0, 0);

        bf16x16 a   = load_a_bf(h_in, H_, m0, kc, lane);
        bf16x16 bb0 = load_b_frag(Wh, H_, n0_0, kc, lane);
        bf16x16 bb1 = load_b_frag(Wh, H_, n0_1, kc, lane);
        acc0 = __builtin_amdgcn_wmma_f32_16x16x32_bf16(false, a, false, bb0, (short)0, acc0, false, false);
        acc1 = __builtin_amdgcn_wmma_f32_16x16x32_bf16(false, a, false, bb1, (short)0, acc1, false, false);
    }

    int h = lane >> 4, n = lane & 15;
#pragma unroll
    for (int r = 0; r < 8; ++r) {
        int m = m0 + h * 8 + r;
        lds_g[g0][m][n] = acc0[r] + Xp_t[(size_t)m * G4H + n0_0 + n];
        lds_g[g1][m][n] = acc1[r] + Xp_t[(size_t)m * G4H + n0_1 + n];
    }
    __syncthreads();

    // elementwise cell update: 64 batch x 16 hidden = 1024 cells, 256 threads
    for (int idx = threadIdx.x; idx < B_ * 16; idx += 256) {
        int b = idx >> 4;
        int j = idx & 15;
        float ig = sigm(lds_g[0][b][j]);
        float fg = sigm(lds_g[1][b][j]);
        float ag = tanh_fast(lds_g[2][b][j]);
        float og = sigm(lds_g[3][b][j]);

        size_t ci = (size_t)b * H_ + j0 + j;
        float c_old = c_ws[ci];
        float cn = ht1(c_old);                 // normalize_hidden_state
        float cy = fg * cn + ig * ag;
        float hy = og * tanh_fast(cy);
        float hy1 = ht1(hy);                   // out_quant == rec_quant

        c_ws[ci]  = cy;
        h_out[ci] = (bf16)hy1;
        out_t[ci] = hy1;
        if (last) { hn_out[ci] = hy1; cn_out[ci] = cy; }
    }
}

// ---------------- host side ----------------
extern "C" void kernel_launch(void* const* d_in, const int* in_sizes, int n_in,
                              void* d_out, int out_size, void* d_ws, size_t ws_size,
                              hipStream_t stream) {
    (void)in_sizes; (void)n_in; (void)out_size; (void)ws_size;

    const float* inputs = (const float*)d_in[0];   // [T, B, I]
    const float* hx     = (const float*)d_in[1];   // [B, H]
    const float* cx     = (const float*)d_in[2];   // [B, H]
    const float* w_ii   = (const float*)d_in[3];
    const float* w_fi   = (const float*)d_in[4];
    const float* w_ai   = (const float*)d_in[5];
    const float* w_oi   = (const float*)d_in[6];
    const float* w_ih   = (const float*)d_in[7];
    const float* w_fh   = (const float*)d_in[8];
    const float* w_ah   = (const float*)d_in[9];
    const float* w_oh   = (const float*)d_in[10];
    const float* b_i    = (const float*)d_in[11];
    const float* b_f    = (const float*)d_in[12];
    const float* b_a    = (const float*)d_in[13];
    const float* b_o    = (const float*)d_in[14];

    float* out = (float*)d_out;                    // outputs [T,B,H] ++ h_n [B,H] ++ c_n [B,H]

    char* ws = (char*)d_ws;
    bf16*  Wi   = (bf16*)(ws + OFF_WI);
    bf16*  Wh   = (bf16*)(ws + OFF_WH);
    float* bias = (float*)(ws + OFF_BIAS);
    bf16*  h0   = (bf16*)(ws + OFF_H0);
    bf16*  h1   = (bf16*)(ws + OFF_H1);
    float* c    = (float*)(ws + OFF_C);
    bf16*  Xbf  = (bf16*)(ws + OFF_XBF);
    float* Xp   = (float*)(ws + OFF_XP);

    // 1) weight conversion + input quantization + state init
    k_cvt4<<<2048, 256, 0, stream>>>(w_ii, w_fi, w_ai, w_oi, Wi, H_, I_);
    k_cvt4<<<2048, 256, 0, stream>>>(w_ih, w_fh, w_ah, w_oh, Wh, H_, H_);
    k_cvt_x<<<4096, 256, 0, stream>>>(inputs, Xbf);
    k_init<<<256, 256, 0, stream>>>(hx, cx, b_i, b_f, b_a, b_o, h0, c, bias);

    // 2) massively parallel input projection for all timesteps
    dim3 g_gemm(G4H / 64, (T_ * B_) / 128);
    k_input_gemm<<<g_gemm, 256, 0, stream>>>(Xbf, Wi, bias, Xp);

    // 3) sequential recurrence (double-buffered hidden state)
    float* hn_out = out + (size_t)T_ * B_ * H_;
    float* cn_out = hn_out + (size_t)B_ * H_;
    for (int t = 0; t < T_; ++t) {
        const bf16* hin = (t & 1) ? h1 : h0;
        bf16* hout      = (t & 1) ? h0 : h1;
        k_lstm_step<<<H_ / 16, 256, 0, stream>>>(
            Wh, Xp + (size_t)t * B_ * G4H, hin, hout, c,
            out + (size_t)t * B_ * H_, hn_out, cn_out, (t == T_ - 1) ? 1 : 0);
    }
}
